// SimpleRNN_39926015983951
// MI455X (gfx1250) — matrix operand order repaired and checked by reading, not compiled, not run on hardware
//
#include <hip/hip_runtime.h>
#include <hip/hip_bf16.h>

#define INPUT_SZ 8
#define HIDDEN   128
#define BATCH    256
#define SEQ      2048
#define CHUNK    64                 // timesteps of x staged in LDS at a time
#define NCH      (SEQ / CHUNK)      // 32 chunks
#define HROW     136                // bf16 row stride for h  (68 dw = 4 mod 64)
#define XROW     520                // bf16 row stride for x  (260 dw = 4 mod 64)

typedef __attribute__((ext_vector_type(16))) __bf16 v16bf;
typedef __attribute__((ext_vector_type(8)))  float  v8f;
typedef __attribute__((ext_vector_type(4)))  __bf16 bf16x4;

union FragU { v16bf v; uint4 q[2]; };

__device__ __forceinline__ float htanh(float v) {
#if __has_builtin(__builtin_amdgcn_tanhf)
    return __builtin_amdgcn_tanhf(v);          // v_tanh_f32 (gfx1250 TRANS op)
#else
    float r;
    asm("v_tanh_f32 %0, %1\n\tv_nop\n\tv_nop" : "=v"(r) : "v"(v));
    return r;
#endif
}

__global__ __launch_bounds__(256) void rnn_scan_kernel(
    const float* __restrict__ x,      // [BATCH][SEQ][8]
    const float* __restrict__ h0,     // [1][BATCH][HIDDEN]
    const float* __restrict__ W_ih,   // [HIDDEN][8]
    const float* __restrict__ W_hh,   // [HIDDEN][HIDDEN]
    const float* __restrict__ b_ih,   // [HIDDEN]
    const float* __restrict__ b_hh,   // [HIDDEN]
    const float* __restrict__ W_head, // [1][HIDDEN]
    const float* __restrict__ b_head, // [1]
    float* __restrict__ out)          // pred[256] | feat[256*128] | h_n[256*128]
{
    const int tid  = threadIdx.x;
    const int lane = tid & 31;
    const int wave = tid >> 5;
    const int g    = lane >> 4;      // half-wave selector
    const int ln   = lane & 15;
    const int n    = wave * 16 + ln; // hidden column owned by this lane
    const int b0   = blockIdx.x * 16;

    float* pred = out;
    float* feat = out + BATCH;
    float* hn   = out + BATCH + BATCH * HIDDEN;

    __shared__ __align__(16) __bf16 h_lds[2][16][HROW];
    __shared__ __align__(16) __bf16 x_bf[2][16][XROW];
    __shared__ float pred_lds[16];

    // ---- per-lane constants (live for the whole scan) ----
    const float bias  = b_ih[n] + b_hh[n];
    const float whead = W_head[n];

    // W_ih B-fragment: B[k][n] = W_ih[n][k], k<8, zero-padded to K=32.
    FragU wih;
    for (int j = 0; j < 16; ++j) wih.v[j] = (__bf16)0.0f;
    if (g == 0)
        for (int j = 0; j < INPUT_SZ; ++j)
            wih.v[j] = (__bf16)W_ih[n * INPUT_SZ + j];

    // W_hh B-fragments (register-resident for the whole scan)
    FragU whh[4];
    for (int c = 0; c < 4; ++c) {
        const float* wrow = W_hh + (size_t)n * HIDDEN + c * 32 + g * 8;
        for (int j = 0; j < 16; ++j) whh[c].v[j] = (__bf16)wrow[j < 8 ? j : j + 8];
    }

    // ---- x staging helpers ----
    const int    xrr = tid >> 4;          // batch row 0..15
    const int    xc  = tid & 15;
    const float* xrow_base = x + (size_t)(b0 + xrr) * SEQ * INPUT_SZ;
    float4 xr[8];

    // chunk 0: synchronous stage (f32 -> bf16 in LDS)
    {
        const float4* src = (const float4*)xrow_base;
        for (int j = 0; j < 8; ++j) xr[j] = src[xc + 16 * j];
        for (int j = 0; j < 8; ++j) {
            float4 f = xr[j];
            bf16x4 b; b[0] = (__bf16)f.x; b[1] = (__bf16)f.y;
                      b[2] = (__bf16)f.z; b[3] = (__bf16)f.w;
            *(bf16x4*)&x_bf[0][xrr][(xc + 16 * j) * 4] = b;
        }
    }

    // ---- initialise h buffer 0 and pred accumulator ----
    for (int e = 0; e < 8; ++e) {
        int idx = tid * 8 + e;
        int m = idx >> 7, k = idx & 127;
        h_lds[0][m][k] = (__bf16)h0[(size_t)(b0 + m) * HIDDEN + k];
    }
    if (tid < 16) pred_lds[tid] = 0.0f;
    __syncthreads();

    for (int tc = 0; tc < NCH; ++tc) {
        // prefetch next x chunk into registers (64 steps of latency to hide)
        if (tc + 1 < NCH) {
            const float4* src = (const float4*)(xrow_base +
                (size_t)(tc + 1) * CHUNK * INPUT_SZ);
            for (int j = 0; j < 8; ++j) xr[j] = src[xc + 16 * j];
        }
        const int xbuf = tc & 1;

        #pragma unroll 2
        for (int dt = 0; dt < CHUNK; ++dt) {
            const int t   = tc * CHUNK + dt;
            const int cur = t & 1, nxt = cur ^ 1;

            // two independent accumulation chains to halve WMMA chain depth
            v8f accA, accB;
            for (int r = 0; r < 8; ++r) accA[r] = bias;
            for (int r = 0; r < 8; ++r) accB[r] = 0.0f;

            // x A-fragment: one b128 LDS load (lanes 0-15), rest zero
            FragU xa;
            xa.q[1] = make_uint4(0u, 0u, 0u, 0u);
            if (g == 0) xa.q[0] = *(const uint4*)&x_bf[xbuf][ln][dt * INPUT_SZ];
            else        xa.q[0] = make_uint4(0u, 0u, 0u, 0u);

            FragU ha0, ha1, ha2, ha3;
            ha0.q[0] = *(const uint4*)&h_lds[cur][ln][0 * 32 + g * 8];
            ha0.q[1] = *(const uint4*)&h_lds[cur][ln][0 * 32 + 16 + g * 8];
            ha1.q[0] = *(const uint4*)&h_lds[cur][ln][1 * 32 + g * 8];
            ha1.q[1] = *(const uint4*)&h_lds[cur][ln][1 * 32 + 16 + g * 8];
            ha2.q[0] = *(const uint4*)&h_lds[cur][ln][2 * 32 + g * 8];
            ha2.q[1] = *(const uint4*)&h_lds[cur][ln][2 * 32 + 16 + g * 8];
            ha3.q[0] = *(const uint4*)&h_lds[cur][ln][3 * 32 + g * 8];
            ha3.q[1] = *(const uint4*)&h_lds[cur][ln][3 * 32 + 16 + g * 8];

            // chain A: bias + x-projection + W_hh chunks 0,1
            accA = __builtin_amdgcn_wmma_f32_16x16x32_bf16(
                false, xa.v,  false, wih.v,    (short)0, accA, false, false);
            accA = __builtin_amdgcn_wmma_f32_16x16x32_bf16(
                false, ha0.v, false, whh[0].v, (short)0, accA, false, false);
            accA = __builtin_amdgcn_wmma_f32_16x16x32_bf16(
                false, ha1.v, false, whh[1].v, (short)0, accA, false, false);
            // chain B: W_hh chunks 2,3 (independent of chain A)
            accB = __builtin_amdgcn_wmma_f32_16x16x32_bf16(
                false, ha2.v, false, whh[2].v, (short)0, accB, false, false);
            accB = __builtin_amdgcn_wmma_f32_16x16x32_bf16(
                false, ha3.v, false, whh[3].v, (short)0, accB, false, false);

            float hv[8];
            for (int r = 0; r < 8; ++r) hv[r] = htanh(accA[r] + accB[r]);

            if (t == SEQ - 1) {
                for (int r = 0; r < 8; ++r) {
                    const int m = r + 8 * g;           // D layout: M = vgpr + 8g
                    const size_t o = (size_t)(b0 + m) * HIDDEN + n;
                    feat[o] = hv[r];
                    hn[o]   = hv[r];
                    atomicAdd(&pred_lds[m], hv[r] * whead);   // ds_add_f32
                }
            }

            for (int r = 0; r < 8; ++r) {
                const int m = r + 8 * g;
                h_lds[nxt][m][n] = (__bf16)hv[r];
            }
            __syncthreads();
        }

        // stage prefetched chunk into the other x buffer (f32 -> bf16)
        if (tc + 1 < NCH) {
            for (int j = 0; j < 8; ++j) {
                float4 f = xr[j];
                bf16x4 b; b[0] = (__bf16)f.x; b[1] = (__bf16)f.y;
                          b[2] = (__bf16)f.z; b[3] = (__bf16)f.w;
                *(bf16x4*)&x_bf[xbuf ^ 1][xrr][(xc + 16 * j) * 4] = b;
            }
        }
        __syncthreads();
    }

    if (tid < 16) pred[b0 + tid] = pred_lds[tid] + b_head[0];
}

extern "C" void kernel_launch(void* const* d_in, const int* in_sizes, int n_in,
                              void* d_out, int out_size, void* d_ws, size_t ws_size,
                              hipStream_t stream) {
    (void)in_sizes; (void)n_in; (void)out_size; (void)d_ws; (void)ws_size;
    const float* x      = (const float*)d_in[0];
    const float* h0     = (const float*)d_in[1];
    const float* W_ih   = (const float*)d_in[2];
    const float* W_hh   = (const float*)d_in[3];
    const float* b_ih   = (const float*)d_in[4];
    const float* b_hh   = (const float*)d_in[5];
    const float* W_head = (const float*)d_in[6];
    const float* b_head = (const float*)d_in[7];
    float* out = (float*)d_out;

    rnn_scan_kernel<<<dim3(BATCH / 16), dim3(256), 0, stream>>>(
        x, h0, W_ih, W_hh, b_ih, b_hh, W_head, b_head, out);
}
